// Model_85916525789266
// MI455X (gfx1250) — matrix operand order repaired and checked by reading, not compiled, not run on hardware
//
#include <hip/hip_runtime.h>
#include <hip/hip_bf16.h>
#include <math.h>

typedef __attribute__((ext_vector_type(16))) _Float16 v16h;
typedef __attribute__((ext_vector_type(8)))  float    v8f;

static constexpr int SEQ = 48;
static constexpr int EMB = 18;
static constexpr int INR = 36;   // I
static constexpr int NHD = 2;
static constexpr int DHM = 18;
static constexpr int FFD = 64;

__device__ __forceinline__ float sigmoidf_(float x) { return 1.f / (1.f + expf(-x)); }
__device__ __forceinline__ float siluf_(float x)    { return x / (1.f + expf(-x)); }
__device__ __forceinline__ float lsigf_(float x) {
  // stable log(sigmoid(x))
  return (x >= 0.f) ? -log1pf(expf(-x)) : (x - log1pf(expf(x)));
}
__device__ __forceinline__ float geluf_(float x) {
  return 0.5f * x * (1.f + erff(x * 0.70710678118654752f));
}

// ---------------------------------------------------------------------------
// mLSTM block: one workgroup (64 threads = 2 waves) per batch element.
// Wave w handles head w in the attention phase using v_wmma_f32_16x16x32_f16.
// All fragment gathers use clamped addresses + value selects so LDS loads are
// unconditional (no exec-masked predication) and EXEC stays all-ones.
// ---------------------------------------------------------------------------
__global__ __launch_bounds__(64) void mlstm_kernel(
    float* __restrict__ x,                 // (B,S,E) in/out
    const float* __restrict__ ln_w,        // (E)
    const float* __restrict__ up_w,        // (E,2I)
    const float* __restrict__ conv_w,      // (I,4)
    const float* __restrict__ conv_b,      // (I)
    const float* __restrict__ q_w,         // (9,4,4)
    const float* __restrict__ k_w,         // (9,4,4)
    const float* __restrict__ v_w,         // (9,4,4)
    const float* __restrict__ ig_w,        // (NH,3I)
    const float* __restrict__ ig_b,        // (NH)
    const float* __restrict__ fg_w,        // (NH,3I)
    const float* __restrict__ fg_b,        // (NH)
    const float* __restrict__ skip_w,      // (I)
    const float* __restrict__ on_w,        // (I)
    const float* __restrict__ down_w)      // (I,E)
{
  __shared__ float s_h [SEQ * EMB];   // LN(x)
  __shared__ float s_xm[SEQ * INR];   // up[:, :I]; reused as ht after attention
  __shared__ float s_z [SEQ * INR];   // up[:, I:]
  __shared__ float s_xc[SEQ * INR];   // silu(conv(xm))
  __shared__ float s_q [SEQ * INR];   // reused as hst at the end
  __shared__ float s_k [SEQ * INR];
  __shared__ float s_v [SEQ * INR];
  __shared__ float s_qk[NHD][SEQ * SEQ];
  __shared__ float s_ipre[NHD][SEQ];
  __shared__ float s_fpre[NHD][SEQ];
  __shared__ float s_lfc [NHD][SEQ];

  const int tid = threadIdx.x;
  float* xb = x + (size_t)blockIdx.x * SEQ * EMB;

  // -------- 1. LayerNorm ------------------------------------------------
  for (int t = tid; t < SEQ; t += 64) {
    float row[EMB];
    float mu = 0.f;
    for (int i = 0; i < EMB; ++i) { row[i] = xb[t * EMB + i]; mu += row[i]; }
    mu *= (1.f / EMB);
    float var = 0.f;
    for (int i = 0; i < EMB; ++i) { float d = row[i] - mu; var += d * d; }
    var *= (1.f / EMB);
    float inv = rsqrtf(var + 1e-5f);
    for (int i = 0; i < EMB; ++i) s_h[t * EMB + i] = (row[i] - mu) * inv * ln_w[i];
  }
  __syncthreads();

  // -------- 2. up projection (18 -> 72) ---------------------------------
  for (int idx = tid; idx < SEQ * 2 * INR; idx += 64) {
    int t = idx / (2 * INR), j = idx % (2 * INR);
    float acc = 0.f;
    for (int i = 0; i < EMB; ++i) acc += s_h[t * EMB + i] * up_w[i * (2 * INR) + j];
    if (j < INR) s_xm[t * INR + j] = acc;
    else         s_z [t * INR + (j - INR)] = acc;
  }
  __syncthreads();

  // -------- 3. causal depthwise conv (K=4) + silu ------------------------
  for (int idx = tid; idx < SEQ * INR; idx += 64) {
    int t = idx / INR, c = idx % INR;
    float acc = conv_b[c];
    for (int kk = 0; kk < 4; ++kk) {
      int ts = t + kk - 3;
      int tc = (ts >= 0) ? ts : 0;                 // clamp address
      float xv = s_xm[tc * INR + c];               // unconditional load
      acc += ((ts >= 0) ? xv : 0.f) * conv_w[c * 4 + kk];
    }
    s_xc[idx] = siluf_(acc);
  }
  __syncthreads();

  // -------- 4. headwise q,k,v (9 groups of 4->4) -------------------------
  for (int idx = tid; idx < SEQ * INR; idx += 64) {
    int t = idx / INR, col = idx % INR;
    int n = col >> 2, o = col & 3;
    float aq = 0.f, ak = 0.f, av = 0.f;
    for (int d = 0; d < 4; ++d) {
      float xcv = s_xc[t * INR + n * 4 + d];
      float xmv = s_xm[t * INR + n * 4 + d];
      float wq = q_w[(n * 4 + o) * 4 + d];
      float wk = k_w[(n * 4 + o) * 4 + d];
      float wv = v_w[(n * 4 + o) * 4 + d];
      aq += xcv * wq; ak += xcv * wk; av += xmv * wv;
    }
    s_q[idx] = aq; s_k[idx] = ak; s_v[idx] = av;
  }
  __syncthreads();

  // -------- 5. gate preactivations ---------------------------------------
  for (int idx = tid; idx < NHD * SEQ; idx += 64) {
    int h = idx / SEQ, t = idx % SEQ;
    float ai = ig_b[h], af = fg_b[h];
    const float* wi = ig_w + h * (3 * INR);
    const float* wf = fg_w + h * (3 * INR);
    for (int f = 0; f < INR; ++f) {
      float qv = s_q[t * INR + f], kv = s_k[t * INR + f], vv = s_v[t * INR + f];
      ai += qv * wi[f] + kv * wi[INR + f] + vv * wi[2 * INR + f];
      af += qv * wf[f] + kv * wf[INR + f] + vv * wf[2 * INR + f];
    }
    s_ipre[h][t] = ai;
    s_fpre[h][t] = af;
  }
  __syncthreads();

  // -------- 6. cumulative log-sigmoid forget gate ------------------------
  if ((tid & 31) == 0) {
    int h = tid >> 5;
    float acc = 0.f;
    for (int t = 0; t < SEQ; ++t) { acc += lsigf_(s_fpre[h][t]); s_lfc[h][t] = acc; }
  }
  __syncthreads();

  // -------- 7. attention: wave w -> head w, WMMA 16x16x32 f16 -------------
  {
    const int hh   = tid >> 5;       // head == wave id
    const int lane = tid & 31;
    const int mrow = lane & 15;
    const int hi   = lane >> 4;
    const float kscale = 0.23570226039551584f; // 18^-0.5
    const int hoff = hh * DHM;

    // GEMM1: qk = Qh (48x18) @ Kh^T -> s_qk[hh] (48x48), K padded 18->32
    for (int tm = 0; tm < 3; ++tm) {
      const float* qrow = &s_q[(tm * 16 + mrow) * INR + hoff];
      v16h a;
#pragma unroll
      for (int j = 0; j < 16; ++j) {
        int kk = (j < 8 ? j : j + 8) + hi * 8;           // A-frag K striping
        int kc = (kk < DHM) ? kk : 0;                    // clamped address
        float raw = qrow[kc];                            // unconditional load
        a[j] = (_Float16)((kk < DHM) ? raw : 0.f);       // value select
      }
      for (int tn = 0; tn < 3; ++tn) {
        const float* krow = &s_k[(tn * 16 + mrow) * INR + hoff];
        v16h bm;
#pragma unroll
        for (int j = 0; j < 16; ++j) {
          int kk = (j < 8 ? j : j + 8) + hi * 8;         // B-frag K striping
          int kc = (kk < DHM) ? kk : 0;
          float raw = krow[kc] * kscale;
          bm[j] = (_Float16)((kk < DHM) ? raw : 0.f);
        }
        v8f c = {};
        c = __builtin_amdgcn_wmma_f32_16x16x32_f16(false, a, false, bm,
                                                   (short)0, c, false, false);
#pragma unroll
        for (int r = 0; r < 8; ++r) {
          int m = tm * 16 + hi * 8 + r;
          int n = tn * 16 + mrow;
          s_qk[hh][m * SEQ + n] = c[r];
        }
      }
    }
    __syncthreads();

    // D-matrix, stabilization, row normalization (in place on s_qk)
    for (int r = lane; r < SEQ; r += 32) {
      float lr = s_lfc[hh][r];
      float maxd = -1e30f;
      for (int t = 0; t <= r; ++t)
        maxd = fmaxf(maxd, lr - s_lfc[hh][t] + s_ipre[hh][t]);
      float sum = 0.f;
      for (int t = 0; t <= r; ++t) {
        float ld = lr - s_lfc[hh][t] + s_ipre[hh][t];
        float Cv = s_qk[hh][r * SEQ + t] * expf(ld - maxd);
        s_qk[hh][r * SEQ + t] = Cv;
        sum += Cv;
      }
      float inv = 1.f / (fmaxf(fabsf(sum), expf(-maxd)) + 1e-6f);
      for (int t = 0; t < SEQ; ++t) {
        float Cv = s_qk[hh][r * SEQ + t];
        s_qk[hh][r * SEQ + t] = (t <= r) ? Cv * inv : 0.f;
      }
    }
    __syncthreads();

    // GEMM2: ht = Cn (48x48) @ Vh (48x18) -> s_xm (reused as ht), K 48->64
    for (int tm = 0; tm < 3; ++tm) {
      const float* arow = &s_qk[hh][(tm * 16 + mrow) * SEQ];
      for (int tn = 0; tn < 2; ++tn) {
        const int nn = tn * 16 + mrow;
        const int nc = (nn < DHM) ? nn : 0;              // clamped N address
        v8f c = {};
        for (int ks = 0; ks < 2; ++ks) {
          v16h a, bm;
#pragma unroll
          for (int j = 0; j < 16; ++j) {
            int ka = ks * 32 + (j < 8 ? j : j + 8) + hi * 8;
            int kac = (ka < SEQ) ? ka : 0;
            float ra = arow[kac];                        // unconditional load
            a[j] = (_Float16)((ka < SEQ) ? ra : 0.f);

            int kb = ks * 32 + (j < 8 ? j : j + 8) + hi * 8;
            int kbc = (kb < SEQ) ? kb : 0;
            float rb = s_v[kbc * INR + hoff + nc];       // unconditional load
            bm[j] = (_Float16)(((kb < SEQ) && (nn < DHM)) ? rb : 0.f);
          }
          c = __builtin_amdgcn_wmma_f32_16x16x32_f16(false, a, false, bm,
                                                     (short)0, c, false, false);
        }
#pragma unroll
        for (int r = 0; r < 8; ++r) {
          int m = tm * 16 + hi * 8 + r;
          if (nn < DHM) s_xm[m * INR + hoff + nn] = c[r];
        }
      }
    }
  }
  __syncthreads();

  // -------- 8. per-head output group-norm --------------------------------
  for (int idx = tid; idx < SEQ * NHD; idx += 64) {
    int t = idx >> 1, h = idx & 1;
    float* row = &s_xm[t * INR + h * DHM];
    float mu = 0.f;
    for (int d = 0; d < DHM; ++d) mu += row[d];
    mu *= (1.f / DHM);
    float var = 0.f;
    for (int d = 0; d < DHM; ++d) { float df = row[d] - mu; var += df * df; }
    var *= (1.f / DHM);
    float inv = rsqrtf(var + 1e-5f);
    for (int d = 0; d < DHM; ++d) row[d] = (row[d] - mu) * inv * on_w[h * DHM + d];
  }
  __syncthreads();

  // -------- 9. hst = (hn + skip*xc) * silu(z) -> s_q (reuse) --------------
  for (int idx = tid; idx < SEQ * INR; idx += 64) {
    s_q[idx] = (s_xm[idx] + skip_w[idx % INR] * s_xc[idx]) * siluf_(s_z[idx]);
  }
  __syncthreads();

  // -------- 10. residual down projection (36 -> 18) ----------------------
  for (int idx = tid; idx < SEQ * EMB; idx += 64) {
    int t = idx / EMB, e = idx % EMB;
    float acc = xb[idx];
    for (int c = 0; c < INR; ++c) acc += s_q[t * INR + c] * down_w[c * EMB + e];
    xb[idx] = acc;
  }
}

// ---------------------------------------------------------------------------
// sLSTM block + fused FFN: one workgroup (64 threads) per batch element.
// Serial scan over 48 steps with 18-wide recurrent state kept in LDS.
// ---------------------------------------------------------------------------
__global__ __launch_bounds__(64) void slstm_kernel(
    float* __restrict__ x,
    const float* __restrict__ ln_w,        // (E)
    const float* __restrict__ conv_w,      // (E,4)
    const float* __restrict__ conv_b,      // (E)
    const float* __restrict__ gate_w,      // (4,NH,9,9)
    const float* __restrict__ rec_w,       // (NH,9,4,9)
    const float* __restrict__ bias,        // (4,NH,9)
    const float* __restrict__ gn_w,        // (E)
    const float* __restrict__ fln_w,       // (E)
    const float* __restrict__ fup_w,       // (E,2*FF)
    const float* __restrict__ fdn_w)       // (FF,E)
{
  __shared__ float s_h [SEQ * EMB];
  __shared__ float s_hc[SEQ * EMB];
  __shared__ float s_wx[SEQ * 4 * EMB];    // [t][g][h*9+e]
  __shared__ float s_y [EMB];
  __shared__ float s_ys[SEQ * EMB];
  __shared__ float s_x2[SEQ * EMB];
  __shared__ float s_ff[SEQ * FFD];

  const int tid = threadIdx.x;
  float* xb = x + (size_t)blockIdx.x * SEQ * EMB;

  // LayerNorm
  for (int t = tid; t < SEQ; t += 64) {
    float row[EMB];
    float mu = 0.f;
    for (int i = 0; i < EMB; ++i) { row[i] = xb[t * EMB + i]; mu += row[i]; }
    mu *= (1.f / EMB);
    float var = 0.f;
    for (int i = 0; i < EMB; ++i) { float d = row[i] - mu; var += d * d; }
    var *= (1.f / EMB);
    float inv = rsqrtf(var + 1e-5f);
    for (int i = 0; i < EMB; ++i) s_h[t * EMB + i] = (row[i] - mu) * inv * ln_w[i];
  }
  __syncthreads();

  // causal conv + silu
  for (int idx = tid; idx < SEQ * EMB; idx += 64) {
    int t = idx / EMB, c = idx % EMB;
    float acc = conv_b[c];
    for (int kk = 0; kk < 4; ++kk) {
      int ts = t + kk - 3;
      int tc = (ts >= 0) ? ts : 0;
      float hv = s_h[tc * EMB + c];
      acc += ((ts >= 0) ? hv : 0.f) * conv_w[c * 4 + kk];
    }
    s_hc[idx] = siluf_(acc);
  }
  __syncthreads();

  // input-gate projections wx[t][g][h,e]  (g<2 from hc, g>=2 from h)
  for (int idx = tid; idx < SEQ * 4 * EMB; idx += 64) {
    int t = idx / (4 * EMB);
    int rem = idx % (4 * EMB);
    int g = rem / EMB, he = rem % EMB;
    int h = he / 9, e = he % 9;
    const float* inp = (g < 2) ? s_hc : s_h;
    float acc = 0.f;
    for (int d = 0; d < 9; ++d)
      acc += inp[t * EMB + h * 9 + d] * gate_w[((g * NHD + h) * 9 + e) * 9 + d];
    s_wx[idx] = acc;
  }
  if (tid < EMB) s_y[tid] = 0.f;
  __syncthreads();

  // recurrent scan
  float cc = 0.f, nc = 0.f, mc = 0.f;
  for (int t = 0; t < SEQ; ++t) {
    float raw[4] = {0.f, 0.f, 0.f, 0.f};
    if (tid < EMB) {
      int h = tid / 9, e = tid % 9;
      for (int g = 0; g < 4; ++g) {
        float acc = s_wx[(t * 4 + g) * EMB + tid] + bias[(g * NHD + h) * 9 + e];
        for (int d = 0; d < 9; ++d)
          acc += s_y[h * 9 + d] * rec_w[((h * 9 + d) * 4 + g) * 9 + e];
        raw[g] = acc;
      }
    }
    __syncthreads();
    if (tid < EMB) {
      float i_ = raw[0], f_ = raw[1], z_ = raw[2], o_ = raw[3];
      float lfm = mc + lsigf_(f_);
      float mn  = fmaxf(i_, lfm);
      float ig  = expf(i_ - mn);
      float fg  = expf(lfm - mn);
      cc = fg * cc + ig * tanhf(z_);
      nc = fg * nc + ig;
      mc = mn;
      float yn = sigmoidf_(o_) * cc / nc;
      s_y[tid] = yn;
      s_ys[t * EMB + tid] = yn;
    }
    __syncthreads();
  }

  // group-norm over DHS=9 per (t,h), residual into s_x2
  for (int idx = tid; idx < SEQ * NHD; idx += 64) {
    int t = idx >> 1, h = idx & 1;
    const float* row = &s_ys[t * EMB + h * 9];
    float mu = 0.f;
    for (int d = 0; d < 9; ++d) mu += row[d];
    mu *= (1.f / 9.f);
    float var = 0.f;
    for (int d = 0; d < 9; ++d) { float df = row[d] - mu; var += df * df; }
    var *= (1.f / 9.f);
    float inv = rsqrtf(var + 1e-5f);
    for (int d = 0; d < 9; ++d) {
      int e = h * 9 + d;
      s_x2[t * EMB + e] = xb[t * EMB + e] + (row[d] - mu) * inv * gn_w[e];
    }
  }
  __syncthreads();

  // FFN: LN(s_x2) -> s_h (reuse)
  for (int t = tid; t < SEQ; t += 64) {
    float row[EMB];
    float mu = 0.f;
    for (int i = 0; i < EMB; ++i) { row[i] = s_x2[t * EMB + i]; mu += row[i]; }
    mu *= (1.f / EMB);
    float var = 0.f;
    for (int i = 0; i < EMB; ++i) { float d = row[i] - mu; var += d * d; }
    var *= (1.f / EMB);
    float inv = rsqrtf(var + 1e-5f);
    for (int i = 0; i < EMB; ++i) s_h[t * EMB + i] = (row[i] - mu) * inv * fln_w[i];
  }
  __syncthreads();

  // gated FFN hidden: gelu(g) * up
  for (int idx = tid; idx < SEQ * FFD; idx += 64) {
    int t = idx / FFD, j = idx % FFD;
    float g = 0.f, u = 0.f;
    for (int i = 0; i < EMB; ++i) {
      float hv = s_h[t * EMB + i];
      g += hv * fup_w[i * (2 * FFD) + j];
      u += hv * fup_w[i * (2 * FFD) + FFD + j];
    }
    s_ff[idx] = geluf_(g) * u;
  }
  __syncthreads();

  // FFN down + residual -> x
  for (int idx = tid; idx < SEQ * EMB; idx += 64) {
    int t = idx / EMB, e = idx % EMB;
    float acc = s_x2[idx];
    for (int j = 0; j < FFD; ++j) acc += s_ff[t * FFD + j] * fdn_w[j * EMB + e];
    xb[idx] = acc;
  }
}

// ---------------------------------------------------------------------------
// Final post-LN + dense(18->1) + zero first 24 positions.
// ---------------------------------------------------------------------------
__global__ void head_kernel(const float* __restrict__ x,
                            const float* __restrict__ pln,
                            const float* __restrict__ dw,
                            const float* __restrict__ db,
                            float* __restrict__ out, int total)
{
  int idx = blockIdx.x * blockDim.x + threadIdx.x;
  if (idx >= total) return;
  int t = idx % SEQ;
  const float* row = x + (size_t)idx * EMB;
  float mu = 0.f;
  for (int i = 0; i < EMB; ++i) mu += row[i];
  mu *= (1.f / EMB);
  float var = 0.f;
  for (int i = 0; i < EMB; ++i) { float d = row[i] - mu; var += d * d; }
  var *= (1.f / EMB);
  float inv = rsqrtf(var + 1e-5f);
  float acc = db[0];
  for (int i = 0; i < EMB; ++i) acc += (row[i] - mu) * inv * pln[i] * dw[i];
  out[idx] = (t < 24) ? 0.f : acc;
}

extern "C" void kernel_launch(void* const* d_in, const int* in_sizes, int n_in,
                              void* d_out, int out_size, void* d_ws, size_t ws_size,
                              hipStream_t stream) {
  (void)n_in; (void)ws_size; (void)out_size;
  const int B = in_sizes[0] / (SEQ * EMB);

  float* x = (float*)d_ws;
  hipMemcpyAsync(x, d_in[0], (size_t)B * SEQ * EMB * sizeof(float),
                 hipMemcpyDeviceToDevice, stream);

  const float* m_ln_w   = (const float*)d_in[1];
  const float* m_up_w   = (const float*)d_in[2];
  const float* m_conv_w = (const float*)d_in[3];
  const float* m_conv_b = (const float*)d_in[4];
  const float* m_q_w    = (const float*)d_in[5];
  const float* m_k_w    = (const float*)d_in[6];
  const float* m_v_w    = (const float*)d_in[7];
  const float* m_ig_w   = (const float*)d_in[8];
  const float* m_ig_b   = (const float*)d_in[9];
  const float* m_fg_w   = (const float*)d_in[10];
  const float* m_fg_b   = (const float*)d_in[11];
  const float* m_skip   = (const float*)d_in[12];
  const float* m_on_w   = (const float*)d_in[13];
  const float* m_down_w = (const float*)d_in[14];
  const float* s_ln_w   = (const float*)d_in[15];
  const float* s_conv_w = (const float*)d_in[16];
  const float* s_conv_b = (const float*)d_in[17];
  const float* s_gate_w = (const float*)d_in[18];
  const float* s_rec_w  = (const float*)d_in[19];
  const float* s_bias   = (const float*)d_in[20];
  const float* s_gn_w   = (const float*)d_in[21];
  const float* f_ln_w   = (const float*)d_in[22];
  const float* f_up_w   = (const float*)d_in[23];
  const float* f_down_w = (const float*)d_in[24];
  const float* post_ln  = (const float*)d_in[25];
  const float* dense_w  = (const float*)d_in[26];
  const float* dense_b  = (const float*)d_in[27];

  int mj = 0;
  for (int blk = 0; blk < 7; ++blk) {
    if (blk == 1) {
      slstm_kernel<<<B, 64, 0, stream>>>(x, s_ln_w, s_conv_w, s_conv_b,
                                         s_gate_w, s_rec_w, s_bias, s_gn_w,
                                         f_ln_w, f_up_w, f_down_w);
    } else {
      mlstm_kernel<<<B, 64, 0, stream>>>(
          x,
          m_ln_w   + mj * EMB,
          m_up_w   + mj * EMB * 2 * INR,
          m_conv_w + mj * INR * 4,
          m_conv_b + mj * INR,
          m_q_w    + mj * 9 * 16,
          m_k_w    + mj * 9 * 16,
          m_v_w    + mj * 9 * 16,
          m_ig_w   + mj * NHD * 3 * INR,
          m_ig_b   + mj * NHD,
          m_fg_w   + mj * NHD * 3 * INR,
          m_fg_b   + mj * NHD,
          m_skip   + mj * INR,
          m_on_w   + mj * INR,
          m_down_w + mj * INR * EMB);
      ++mj;
    }
  }

  int total = B * SEQ;
  head_kernel<<<(total + 255) / 256, 256, 0, stream>>>(
      x, post_ln, dense_w, dense_b, (float*)d_out, total);
}
